// LSTM_41343355191663
// MI455X (gfx1250) — matrix operand order repaired and checked by reading, not compiled, not run on hardware
//
#include <hip/hip_runtime.h>
#include <hip/hip_bf16.h>

// Problem constants
#define B_   64
#define T_   256
#define I_   256
#define H_   512
#define G4   2048   // 4*H
#define BT   16384  // B*T
#define C_   1000
#define KFC  131072 // T*H (FC reduction dim)

#define STEP_BLOCKS 32   // one uTile per workgroup; grid barrier arrivals

// ---------------------------------------------------------------- types
typedef __attribute__((ext_vector_type(16))) __bf16 bf16x16;
typedef __attribute__((ext_vector_type(8)))  __bf16 bf16x8;
typedef __attribute__((ext_vector_type(8)))  float  f32x8;
typedef __attribute__((ext_vector_type(4)))  float  f32x4;

union FragBF {
    bf16x16 v;
    bf16x8  h[2];
};

__device__ inline f32x8 zero8() {
    f32x8 z;
#pragma unroll
    for (int i = 0; i < 8; ++i) z[i] = 0.0f;
    return z;
}

// A fragment (16x32 bf16, M x K): lane = M row (lane&15).
// lanes 0-15 hold K = k0..k0+15; lanes 16-31 hold K = k0+16..k0+31.
__device__ inline bf16x16 load_a_frag(const __bf16* row, int k0, int hi) {
    FragBF f;
    f.h[0] = *reinterpret_cast<const bf16x8*>(row + k0 + hi * 16);
    f.h[1] = *reinterpret_cast<const bf16x8*>(row + k0 + hi * 16 + 8);
    return f.v;
}

// B fragment (32x16 bf16, K x N): lane = N col (lane&15).
// lanes 0-15 hold K = k0..k0+15; lanes 16-31 hold K = k0+16..k0+31.
__device__ inline bf16x16 load_b_frag(const __bf16* col, int k0, int hi) {
    FragBF f;
    const __bf16* p = col + k0 + hi * 16;
    f.h[0] = *reinterpret_cast<const bf16x8*>(p);
    f.h[1] = *reinterpret_cast<const bf16x8*>(p + 8);
    return f.v;
}

__device__ inline f32x8 wmma_bf16(bf16x16 a, bf16x16 b, f32x8 c) {
    return __builtin_amdgcn_wmma_f32_16x16x32_bf16(
        /*neg_a=*/false, a, /*neg_b=*/false, b,
        /*c_mod=*/(short)0, c, /*reuse_a=*/false, /*reuse_b=*/false);
}

__device__ inline float sigf(float x) { return 1.0f / (1.0f + __expf(-x)); }

// Device-wide split barrier: monotonically increasing counter (no reset races).
__device__ inline void grid_sync(int* cnt, int target) {
    __threadfence();   // make this block's global writes visible (agent scope)
    __syncthreads();
    if (threadIdx.x == 0) {
        __hip_atomic_fetch_add(cnt, 1, __ATOMIC_RELEASE, __HIP_MEMORY_SCOPE_AGENT);
        while (__hip_atomic_load(cnt, __ATOMIC_ACQUIRE, __HIP_MEMORY_SCOPE_AGENT) < target)
            __builtin_amdgcn_s_sleep(1);
    }
    __syncthreads();
}

// ---------------------------------------------------------------- kernels

// fp32 -> bf16 elementwise convert
__global__ void cvt_bf16_kernel(const float* __restrict__ in,
                                __bf16* __restrict__ out, int n) {
    int i = blockIdx.x * blockDim.x + threadIdx.x;
    if (i < n) out[i] = (__bf16)in[i];
}

// zero h (double buffer), c, and the grid-barrier counter
__global__ void init_state_kernel(__bf16* __restrict__ h, float* __restrict__ c,
                                  int* __restrict__ bar) {
    int i = blockIdx.x * blockDim.x + threadIdx.x;
    if (i < 2 * B_ * H_) h[i] = (__bf16)0.0f;
    if (i < B_ * H_)     c[i] = 0.0f;
    if (i == 0)          *bar = 0;
}

// xg = A(BT x K, bf16) * Wih(G4 x K, bf16)^T + b_ih + b_hh   -> fp32 [BT x G4]
// one wave = 16M x 64N strip (4 accumulators sharing the A fragment)
__global__ void gemm_xg_kernel(const __bf16* __restrict__ A,
                               const __bf16* __restrict__ Wih,
                               const float* __restrict__ b_ih,
                               const float* __restrict__ b_hh,
                               float* __restrict__ xg, int K) {
    const int w     = (blockIdx.x * blockDim.x + threadIdx.x) >> 5;
    const int lane  = threadIdx.x & 31;
    const int hi    = lane >> 4;
    const int lm    = lane & 15;
    const int mTile = w >> 5;        // 0..1023
    const int n0    = (w & 31) * 64; // 32 groups of 64 cols

    const __bf16* arow = A + (size_t)(mTile * 16 + lm) * K;
    f32x8 acc[4];
#pragma unroll
    for (int j = 0; j < 4; ++j) acc[j] = zero8();

    for (int k0 = 0; k0 < K; k0 += 32) {
        bf16x16 a = load_a_frag(arow, k0, hi);
#pragma unroll
        for (int j = 0; j < 4; ++j) {
            const __bf16* bcol = Wih + (size_t)(n0 + j * 16 + lm) * K;
            acc[j] = wmma_bf16(a, load_b_frag(bcol, k0, hi), acc[j]);
        }
    }
#pragma unroll
    for (int j = 0; j < 4; ++j) {
        const int  n    = n0 + j * 16 + lm;
        const float bia = b_ih[n] + b_hh[n];
#pragma unroll
        for (int r = 0; r < 8; ++r) {
            const int m = mTile * 16 + r + hi * 8;
            xg[(size_t)m * G4 + n] = acc[j][r] + bia;
        }
    }
}

// Persistent per-layer recurrence. Grid = 32 blocks x 128 threads (4 waves).
// Block b owns hidden-unit tile uTile = b; wave w (=mTile 0..3) computes the
// 4 gate tiles (i,f,g,o) for (mTile, uTile), sharing one A fragment of h.
// W_hh slice for this block (4 gates x 16 rows x 512 = 64 KB) is staged in
// LDS once and served from ds_load for all 256 timesteps. Steps are separated
// by a device-scope atomic barrier (release/acquire) instead of 256 launches.
__global__ void lstm_layer_kernel(const float* __restrict__ xg,
                                  const __bf16* __restrict__ Whh,
                                  __bf16* __restrict__ hbuf,   // 2 x [B_*H_]
                                  float* __restrict__ c,
                                  __bf16* __restrict__ y,
                                  int* __restrict__ bar) {
    extern __shared__ __bf16 Blds[];   // [4*16 rows][512] = 64 KB

    const int lane  = threadIdx.x & 31;
    const int hi    = lane >> 4;
    const int lm    = lane & 15;
    const int mTile = threadIdx.x >> 5;   // 4 waves = 4 batch tiles
    const int uTile = blockIdx.x;         // 0..31

    // Stage W_hh rows {q*512 + uTile*16 + rr : q=0..3, rr=0..15} into LDS.
    // LDS row (q*16+rr) holds that W_hh row (512 bf16, K-contiguous).
    for (int idx = threadIdx.x; idx < (64 * 512) / 8; idx += blockDim.x) {
        const int e   = idx * 8;
        const int row = e >> 9;          // 0..63
        const int col = e & 511;
        const int q   = row >> 4;
        const int rr  = row & 15;
        *reinterpret_cast<bf16x8*>(Blds + (size_t)row * 512 + col) =
            *reinterpret_cast<const bf16x8*>(
                Whh + ((size_t)(q * H_ + uTile * 16 + rr)) * H_ + col);
    }
    __syncthreads();

    const int u = uTile * 16 + lm;

    for (int t = 0; t < T_; ++t) {
        const __bf16* hr = hbuf + (t & 1) * (B_ * H_);        // read t-1 state
        __bf16*       hw = hbuf + ((t + 1) & 1) * (B_ * H_);  // write t state

        const __bf16* arow = hr + (size_t)(mTile * 16 + lm) * H_;
        f32x8 acc[4];
#pragma unroll
        for (int q = 0; q < 4; ++q) acc[q] = zero8();

#pragma unroll 4
        for (int k0 = 0; k0 < H_; k0 += 32) {
            bf16x16 a = load_a_frag(arow, k0, hi);
#pragma unroll
            for (int q = 0; q < 4; ++q) {  // gate order i,f,g,o
                const __bf16* bcol = Blds + (size_t)(q * 16 + lm) * H_;
                acc[q] = wmma_bf16(a, load_b_frag(bcol, k0, hi), acc[q]);
            }
        }

#pragma unroll
        for (int r = 0; r < 8; ++r) {
            const int    m  = mTile * 16 + r + hi * 8;          // batch index
            const size_t xr = ((size_t)m * T_ + t) * G4;        // row b*T+t
            const float ip = acc[0][r] + xg[xr + 0 * H_ + u];
            const float fp = acc[1][r] + xg[xr + 1 * H_ + u];
            const float gp = acc[2][r] + xg[xr + 2 * H_ + u];
            const float op = acc[3][r] + xg[xr + 3 * H_ + u];
            const size_t su = (size_t)m * H_ + u;
            const float cn = sigf(fp) * c[su] + sigf(ip) * tanhf(gp);
            const float hn = sigf(op) * tanhf(cn);
            c[su]  = cn;
            hw[su] = (__bf16)hn;
            y[((size_t)m * T_ + t) * H_ + u] = (__bf16)hn;
        }

        grid_sync(bar, STEP_BLOCKS * (t + 1));
    }
}

// FC partials: out_part[s][m][c] = sum_{k in seg s} flat_bf16[m,k] * fc_w[c,k]
// fc_w is fp32, converted to bf16 in-register (streamed once from HBM).
#define FC_SEGS 16
#define FC_NT   63   // ceil(1000/16)
__global__ void fc_partial_kernel(const __bf16* __restrict__ flat,
                                  const float* __restrict__ fcw,
                                  float* __restrict__ part) {
    const int w     = (blockIdx.x * blockDim.x + threadIdx.x) >> 5;
    const int lane  = threadIdx.x & 31;
    const int hi    = lane >> 4;
    const int lm    = lane & 15;
    const int seg   = w / (4 * FC_NT);
    const int rem   = w % (4 * FC_NT);
    const int mTile = rem / FC_NT;
    const int nTile = rem % FC_NT;

    const __bf16* arow  = flat + (size_t)(mTile * 16 + lm) * KFC;
    const int     col   = nTile * 16 + lm;
    const bool    valid = (col < C_);
    const float*  wrow  = fcw + (size_t)(valid ? col : 0) * KFC;

    f32x8 acc = zero8();
    const int kBeg = seg * (KFC / FC_SEGS);
    const int kEnd = kBeg + (KFC / FC_SEGS);
    for (int k0 = kBeg; k0 < kEnd; k0 += 32) {
        bf16x16 a = load_a_frag(arow, k0, hi);
        FragBF  f;
        const int base = k0 + hi * 16;
#pragma unroll
        for (int i = 0; i < 4; ++i) {
            f32x4 v = *reinterpret_cast<const f32x4*>(wrow + base + i * 4);
#pragma unroll
            for (int j = 0; j < 4; ++j)
                f.v[i * 4 + j] = valid ? (__bf16)v[j] : (__bf16)0.0f;
        }
        acc = wmma_bf16(a, f.v, acc);
    }
#pragma unroll
    for (int r = 0; r < 8; ++r) {
        const int m = mTile * 16 + r + hi * 8;
        part[((size_t)seg * B_ + m) * (FC_NT * 16) + nTile * 16 + lm] = acc[r];
    }
}

__global__ void fc_reduce_kernel(const float* __restrict__ part,
                                 const float* __restrict__ fc_b,
                                 float* __restrict__ out) {
    int i = blockIdx.x * blockDim.x + threadIdx.x;
    if (i >= B_ * C_) return;
    const int b = i / C_, cc = i % C_;
    float s = fc_b[cc];
#pragma unroll
    for (int sgm = 0; sgm < FC_SEGS; ++sgm)
        s += part[((size_t)sgm * B_ + b) * (FC_NT * 16) + cc];
    out[i] = s;
}

// ---------------------------------------------------------------- launcher
extern "C" void kernel_launch(void* const* d_in, const int* in_sizes, int n_in,
                              void* d_out, int out_size, void* d_ws, size_t ws_size,
                              hipStream_t stream) {
    (void)in_sizes; (void)n_in; (void)out_size; (void)ws_size;
    char* ws = (char*)d_ws;

    // Workspace layout (~185 MB)
    float*  xg    = (float*)(ws + 0);                        // BT*G4 fp32   134 MB
    __bf16* xbuf  = (__bf16*)(ws + 134217728);               // BT*H bf16     16 MB
    __bf16* yA    = (__bf16*)(ws + 150994944);               // BT*H bf16     16 MB
    __bf16* yB    = (__bf16*)(ws + 167772160);               // BT*H bf16     16 MB
    __bf16* wih   = (__bf16*)(ws + 184549376);               // G4*H bf16      2 MB
    __bf16* whh   = (__bf16*)(ws + 186646528);               // G4*H bf16      2 MB
    __bf16* hbuf  = (__bf16*)(ws + 188743680);               // 2*B*H bf16   128 KB
    float*  cbuf  = (float*)(ws + 188874752);                // B*H fp32     128 KB
    float*  fcp   = (float*)(ws + 189005824);                // 16*64*1008 fp32 ~4 MB
    int*    bar   = (int*)(ws + 193134592);                  // barrier counter

    const float* x    = (const float*)d_in[0];
    const float* fc_w = (const float*)d_in[13];
    const float* fc_b = (const float*)d_in[14];
    float* out = (float*)d_out;

    // x (fp32) -> bf16 [BT x I]
    cvt_bf16_kernel<<<(BT * I_ + 255) / 256, 256, 0, stream>>>(x, xbuf, BT * I_);

    const __bf16* inbuf = xbuf;
    int inK = I_;
    __bf16* ybufs[2] = { yA, yB };

    for (int l = 0; l < 3; ++l) {
        const float* Wih  = (const float*)d_in[1 + 4 * l];
        const float* Whh  = (const float*)d_in[2 + 4 * l];
        const float* b_ih = (const float*)d_in[3 + 4 * l];
        const float* b_hh = (const float*)d_in[4 + 4 * l];

        cvt_bf16_kernel<<<(G4 * inK + 255) / 256, 256, 0, stream>>>(Wih, wih, G4 * inK);
        cvt_bf16_kernel<<<(G4 * H_  + 255) / 256, 256, 0, stream>>>(Whh, whh, G4 * H_);

        // xg = in @ Wih^T + b_ih + b_hh : 32768 waves
        gemm_xg_kernel<<<4096, 256, 0, stream>>>(inbuf, wih, b_ih, b_hh, xg, inK);

        init_state_kernel<<<256, 256, 0, stream>>>(hbuf, cbuf, bar);

        __bf16* yout = ybufs[l & 1];
        // Persistent recurrence: one launch per layer, 64 KB dynamic LDS/block
        lstm_layer_kernel<<<STEP_BLOCKS, 128, 65536, stream>>>(
            xg, whh, hbuf, cbuf, yout, bar);

        inbuf = yout;
        inK = H_;
    }

    // FC on layer-2 output (== yA, flat [B x T*H] row-major)
    fc_partial_kernel<<<504, 256, 0, stream>>>(inbuf, fc_w, fcp);
    fc_reduce_kernel<<<(B_ * C_ + 255) / 256, 256, 0, stream>>>(fcp, fc_b, out);
}
